// PrototypeHead_24979529794134
// MI455X (gfx1250) — matrix operand order repaired and checked by reading, not compiled, not run on hardware
//
#include <hip/hip_runtime.h>
#include <math.h>

#define BATCH 64
#define CH    768
#define TOK   1024
#define KP    800
#define NCLS  100
#define PPC   8

typedef __attribute__((ext_vector_type(16))) _Float16 v16h;
typedef __attribute__((ext_vector_type(8)))  _Float16 v8h;
typedef __attribute__((ext_vector_type(8)))  float    v8f;

union VU { v16h v; v8h h[2]; };

__device__ __forceinline__ float get_temp(const float* lt) {
    return fminf(fmaxf(__expf(lt[0]), 1e-4f), 100.0f);
}

// ---- K0: normalize prototypes, convert to f16 -------------------------------
__global__ void k_proto(const float* __restrict__ protos, _Float16* __restrict__ ph) {
    __shared__ float red[256];
    int r = blockIdx.x;
    const float* row = protos + (size_t)r * CH;
    float ss = 0.f;
    for (int i = threadIdx.x; i < CH; i += 256) { float v = row[i]; ss += v * v; }
    red[threadIdx.x] = ss; __syncthreads();
    for (int s = 128; s > 0; s >>= 1) {
        if (threadIdx.x < s) red[threadIdx.x] += red[threadIdx.x + s];
        __syncthreads();
    }
    float inv = 1.f / fmaxf(sqrtf(red[0]), 1e-6f);
    for (int i = threadIdx.x; i < CH; i += 256)
        ph[(size_t)r * CH + i] = (_Float16)(row[i] * inv);
}

// ---- K1: per-token inverse L2 norm (coalesced over T) -----------------------
__global__ void k_tnorm(const float* __restrict__ feats, float* __restrict__ invn) {
    int g = blockIdx.x * 256 + threadIdx.x;           // 0 .. 65535
    int b = g >> 10, t = g & 1023;
    const float* p = feats + (size_t)b * CH * TOK + t;
    float ss = 0.f;
    #pragma unroll 4
    for (int c = 0; c < CH; c++) { float v = p[(size_t)c * TOK]; ss += v * v; }
    invn[g] = 1.f / fmaxf(sqrtf(ss), 1e-6f);
}

// ---- K2: LDS-tiled transpose [B,C,T] -> [B,T,C], scale, f16 convert ---------
__global__ void k_transpose(const float* __restrict__ feats,
                            const float* __restrict__ invn,
                            _Float16* __restrict__ th) {
    __shared__ float tile[32][33];
    int b = blockIdx.z, tt = blockIdx.x, ct = blockIdx.y;
    int lt  = threadIdx.x & 31;
    int grp = threadIdx.x >> 5;                        // 0..7
    const float* fb = feats + (size_t)b * CH * TOK;
    #pragma unroll
    for (int i = 0; i < 4; i++) {
        int c = grp + i * 8;                           // local channel
        tile[c][lt] = fb[(size_t)(ct * 32 + c) * TOK + tt * 32 + lt];
    }
    __syncthreads();
    #pragma unroll
    for (int i = 0; i < 4; i++) {
        int tl = grp + i * 8;                          // local token
        int t  = tt * 32 + tl;
        float sc = invn[b * TOK + t];
        th[((size_t)(b * TOK + t)) * CH + ct * 32 + lt] =
            (_Float16)(tile[lt][tl] * sc);
    }
}

// ---- K3: zero evidence accumulators ----------------------------------------
__global__ void k_zero(float* __restrict__ s, int n) {
    int i = blockIdx.x * 256 + threadIdx.x;
    if (i < n) s[i] = 0.f;
}

// ---- K4: WMMA GEMM fused with exp-sum over T -------------------------------
// grid (50 ktiles, 64 batches), block 256 = 8 waves; wave owns 128 tokens x 16 protos.
// Double-buffered K-loop: loads for step ks+1 are in flight while WMMAs consume ks.
__global__ void __launch_bounds__(256)
k_gemm(const _Float16* __restrict__ tok, const _Float16* __restrict__ ph,
       const float* __restrict__ lt, float* __restrict__ sums) {
    int ktile = blockIdx.x, b = blockIdx.y;
    int wave = threadIdx.x >> 5, lane = threadIdx.x & 31;
    int row = lane & 15, hi = lane >> 4;
    float temp = get_temp(lt);

    int tbase = wave * 128;
    // A tile tt, kstep ks: elements 0..7 = ch c0+hi*8.., elements 8..15 = ch c0+16+hi*8..
    const _Float16* aBase = tok + ((size_t)b * TOK + tbase + row) * CH + hi * 8;
    // B tile: lane n = row -> proto (ktile*16+n); elements 0..7 = ch c0+hi*8.., elements 8..15 = ch c0+16+hi*8..
    const _Float16* bBase = ph + (size_t)(ktile * 16 + row) * CH + hi * 8;

    v8f acc[8];
    v8f zc = {};
    #pragma unroll
    for (int i = 0; i < 8; i++) acc[i] = zc;

    VU A0[8], A1[8], B0, B1;

    auto ld = [&](VU (&A)[8], VU& B, int ks) {
        int c0 = ks * 32;
        B.h[0] = *(const v8h*)(bBase + c0);
        B.h[1] = *(const v8h*)(bBase + c0 + 16);
        #pragma unroll
        for (int tt = 0; tt < 8; tt++) {
            const _Float16* ap = aBase + (size_t)tt * 16 * CH + c0;
            A[tt].h[0] = *(const v8h*)(ap);
            A[tt].h[1] = *(const v8h*)(ap + 16);
        }
    };
    auto mma = [&](VU (&A)[8], VU& B) {
        #pragma unroll
        for (int tt = 0; tt < 8; tt++)
            acc[tt] = __builtin_amdgcn_wmma_f32_16x16x32_f16(
                false, A[tt].v, false, B.v, (short)0, acc[tt], false, false);
    };

    ld(A0, B0, 0);                      // prologue
    for (int ks = 0; ks < 22; ks += 2) {
        ld(A1, B1, ks + 1);             // prefetch odd step
        mma(A0, B0);                    // consume even step (overlaps with loads)
        ld(A0, B0, ks + 2);             // prefetch next even step
        mma(A1, B1);                    // consume odd step
    }
    ld(A1, B1, 23);                     // epilogue
    mma(A0, B0);
    mma(A1, B1);

    // D layout: lanes 0-15 N=lane, M=j; lanes 16-31 N=lane-16, M=8+j.
    // evidence term: exp(temp*dot - temp)   (exact LSE shift since dot <= 1)
    float s = 0.f;
    #pragma unroll
    for (int tt = 0; tt < 8; tt++) {
        #pragma unroll
        for (int j = 0; j < 8; j++)
            s += __expf(temp * acc[tt][j] - temp);
    }
    s += __shfl_xor(s, 16, 32);   // combine both M-halves for same proto column
    if (lane < 16)
        atomicAdd(&sums[b * KP + ktile * 16 + lane], s);
}

// ---- K5: logits = temp + log( sum over 8 protos of S ) ----------------------
__global__ void k_final(const float* __restrict__ sums, const float* __restrict__ lt,
                        float* __restrict__ out) {
    int i = blockIdx.x * 256 + threadIdx.x;
    if (i >= BATCH * NCLS) return;
    int b = i / NCLS, cls = i % NCLS;
    float ssum = 0.f;
    #pragma unroll
    for (int p = 0; p < PPC; p++) ssum += sums[b * KP + cls * PPC + p];
    out[i] = get_temp(lt) + logf(ssum);
}

extern "C" void kernel_launch(void* const* d_in, const int* in_sizes, int n_in,
                              void* d_out, int out_size, void* d_ws, size_t ws_size,
                              hipStream_t stream) {
    const float* feats    = (const float*)d_in[0];
    const float* protos   = (const float*)d_in[1];
    const float* log_temp = (const float*)d_in[2];

    char* ws = (char*)d_ws;
    const size_t TOK_BYTES  = (size_t)BATCH * TOK * CH * sizeof(_Float16); // 100663296
    const size_t PROT_BYTES = (size_t)KP * CH * sizeof(_Float16);          // 1228800
    const size_t INVN_BYTES = (size_t)BATCH * TOK * sizeof(float);         // 262144
    _Float16* tokh = (_Float16*)ws;
    _Float16* ph   = (_Float16*)(ws + TOK_BYTES);
    float*    invn = (float*)(ws + TOK_BYTES + PROT_BYTES);
    float*    sums = (float*)(ws + TOK_BYTES + PROT_BYTES + INVN_BYTES);
    float*    out  = (float*)d_out;

    hipLaunchKernelGGL(k_proto,     dim3(KP),            dim3(256), 0, stream, protos, ph);
    hipLaunchKernelGGL(k_tnorm,     dim3(BATCH*TOK/256), dim3(256), 0, stream, feats, invn);
    hipLaunchKernelGGL(k_transpose, dim3(32, 24, BATCH), dim3(256), 0, stream, feats, invn, tokh);
    hipLaunchKernelGGL(k_zero,      dim3((BATCH*KP+255)/256), dim3(256), 0, stream, sums, BATCH*KP);
    hipLaunchKernelGGL(k_gemm,      dim3(KP/16, BATCH),  dim3(256), 0, stream, tokh, ph, log_temp, sums);
    hipLaunchKernelGGL(k_final,     dim3((BATCH*NCLS+255)/256), dim3(256), 0, stream, sums, log_temp, out);
}
